// GRUMultiLayer_17446157156706
// MI455X (gfx1250) — matrix operand order repaired and checked by reading, not compiled, not run on hardware
//
#include <hip/hip_runtime.h>
#include <hip/hip_bf16.h>
#include <stdint.h>
#include <stddef.h>

// Problem constants (from reference): T=2048, D=1024, 3D=3072, L=4
#define TT   2048
#define DD   1024
#define ND3  3072
#define LL   4

typedef __attribute__((ext_vector_type(16))) __bf16 v16bf;
typedef __attribute__((ext_vector_type(8)))  float  v8f;
typedef __attribute__((ext_vector_type(4)))  int    v4i;

// gfx1250 async global->LDS copy builtins (guarded; fallback compiles anywhere)
#if defined(__has_builtin)
#if __has_builtin(__builtin_amdgcn_global_load_async_to_lds_b128) && \
    __has_builtin(__builtin_amdgcn_s_wait_asynccnt)
#define USE_ASYNC_LDS 1
#endif
#endif

#ifdef USE_ASYNC_LDS
// Builtin signature (from compiler diagnostic): param0 = v4i in AS(1)
// ("__device__"), param1 = v4i in AS(3), then imm offset + imm cpol.
typedef __attribute__((address_space(1))) v4i v4i_g;
typedef __attribute__((address_space(3))) v4i v4i_l;
static __device__ __forceinline__ void async_copy16(const void* gsrc, void* ldst) {
  __builtin_amdgcn_global_load_async_to_lds_b128(
      (v4i_g*)(void*)gsrc, (v4i_l*)ldst, 0, 0);
}
#endif

// ---------------------------------------------------------------------------
// fp32 -> bf16 (round-to-nearest-even)
// ---------------------------------------------------------------------------
static __device__ __forceinline__ unsigned short f32_to_bf16_rne(float f) {
  union { float f; unsigned u; } v; v.f = f;
  unsigned u = v.u;
  unsigned r = u + 0x7FFFu + ((u >> 16) & 1u);
  if ((u & 0x7F800000u) == 0x7F800000u) r = u;  // inf/nan passthrough
  return (unsigned short)(r >> 16);
}

__global__ void cvt_f32_bf16_kernel(const float* __restrict__ in,
                                    unsigned short* __restrict__ out, int n) {
  int i = blockIdx.x * blockDim.x + threadIdx.x;
  if (i < n) out[i] = f32_to_bf16_rne(in[i]);
}

// ---------------------------------------------------------------------------
// Wi (L x 1024 x 3072 fp32) -> WiT16 (L x 3072 x 1024 bf16): fused
// transpose + convert so the GEMM B-tile staging is a contiguous copy.
// ---------------------------------------------------------------------------
__global__ __launch_bounds__(256)
void cvt_transpose_wi_kernel(const float* __restrict__ Wi,
                             unsigned short* __restrict__ WiT16) {
  __shared__ float tile[32][33];
  const int l  = blockIdx.z;
  const int kb = blockIdx.y * 32;   // k dim (1024)
  const int nb = blockIdx.x * 32;   // n dim (3072)
  const int tx = threadIdx.x & 31;
  const int ty = threadIdx.x >> 5;  // 0..7
  const float* src = Wi + (size_t)l * DD * ND3;
  unsigned short* dst = WiT16 + (size_t)l * ND3 * DD;
  for (int r = ty; r < 32; r += 8)
    tile[r][tx] = src[(size_t)(kb + r) * ND3 + nb + tx];
  __syncthreads();
  for (int r = ty; r < 32; r += 8)
    dst[(size_t)(nb + r) * DD + kb + tx] = f32_to_bf16_rne(tile[tx][r]);
}

// ---------------------------------------------------------------------------
// Wh (1024 x 3072) -> WhT (3072 x 1024), LDS-tiled transpose (fp32)
// ---------------------------------------------------------------------------
__global__ __launch_bounds__(256)
void transpose_wh_kernel(const float* __restrict__ Wh, float* __restrict__ WhT) {
  __shared__ float tile[32][33];
  const int kb = blockIdx.y * 32;
  const int cb = blockIdx.x * 32;
  const int tx = threadIdx.x & 31;
  const int ty = threadIdx.x >> 5;
  for (int r = ty; r < 32; r += 8)
    tile[r][tx] = Wh[(size_t)(kb + r) * ND3 + cb + tx];
  __syncthreads();
  for (int r = ty; r < 32; r += 8)
    WhT[(size_t)(cb + r) * DD + kb + tx] = tile[tx][r];
}

// ---------------------------------------------------------------------------
// bf16 WMMA GEMM: C[T x 3072] = A[T x 1024] * B[1024 x 3072], fp32 accum.
// B supplied pre-transposed (BT: 3072 x 1024) so both LDS tiles stage as
// contiguous 16B copies (async global->LDS when available).
// Block tile 64x64, K-step 32, 8 waves (wave32), each wave owns a 16x32 C slab.
// ---------------------------------------------------------------------------
__global__ __launch_bounds__(256)
void gemm_xp_kernel(const unsigned short* __restrict__ A,    // bf16, T x 1024
                    const unsigned short* __restrict__ BT,   // bf16, 3072 x 1024
                    float* __restrict__ C) {                 // f32,  T x 3072
  __shared__ __align__(16) unsigned short As[64 * 32];  // [m][k]
  __shared__ __align__(16) unsigned short Bs[64 * 32];  // [n][k]

  const int tid  = threadIdx.x;
  const int lane = tid & 31;
  const int wave = tid >> 5;          // 0..7
  const int m0   = blockIdx.y * 64;
  const int n0   = blockIdx.x * 64;
  const int mi   = wave & 3;          // row sub-tile 0..3
  const int ni2  = wave >> 2;         // 0..1 -> n sub-tiles {2*ni2, 2*ni2+1}

  // Per-thread staging coordinates: one 16B chunk per tile
  const int srow = tid >> 2;          // 0..63
  const int sck  = (tid & 3) * 8;     // 0, 8, 16, 24

  v8f acc0 = {}; v8f acc1 = {};

  for (int k0 = 0; k0 < DD; k0 += 32) {
    const unsigned short* ga = A  + (size_t)(m0 + srow) * DD + k0 + sck;
    const unsigned short* gb = BT + (size_t)(n0 + srow) * DD + k0 + sck;
    unsigned short* la = As + srow * 32 + sck;
    unsigned short* lb = Bs + srow * 32 + sck;
#ifdef USE_ASYNC_LDS
    async_copy16(ga, la);
    async_copy16(gb, lb);
    __builtin_amdgcn_s_wait_asynccnt(0);
#else
    *(uint4*)la = *(const uint4*)ga;
    *(uint4*)lb = *(const uint4*)gb;
#endif
    __syncthreads();

    // A fragment: lane<16 -> K {0..15},
    // lane>=16 -> K {16..31}, row = mi*16 + (lane&15)
    v16bf a;
    {
      const int r  = mi * 16 + (lane & 15);
      const int kb = (lane >> 4) * 16;
      const __bf16* p0 = (const __bf16*)(As + r * 32 + kb);
      const __bf16* p1 = (const __bf16*)(As + r * 32 + kb + 8);
#pragma unroll
      for (int i = 0; i < 8; ++i) { a[i] = p0[i]; a[8 + i] = p1[i]; }
    }
    // B fragments: lane holds 16 consecutive K of column (lane&15);
    // lanes>=16 hold K 16..31. Bs is [n][k] so this is contiguous.
#pragma unroll
    for (int j = 0; j < 2; ++j) {
      const int nl = (ni2 * 2 + j) * 16 + (lane & 15);
      const int kb = (lane >> 4) * 16;
      const __bf16* p = (const __bf16*)(Bs + nl * 32 + kb);
      v16bf b;
#pragma unroll
      for (int i = 0; i < 16; ++i) b[i] = p[i];
      if (j == 0)
        acc0 = __builtin_amdgcn_wmma_f32_16x16x32_bf16(false, a, false, b,
                                                       (short)0, acc0, false, false);
      else
        acc1 = __builtin_amdgcn_wmma_f32_16x16x32_bf16(false, a, false, b,
                                                       (short)0, acc1, false, false);
    }
    __syncthreads();
  }

  // Store C per ISA f32 C/D layout: VGPR r -> row M=r (lanes 0-15), M=8+r (16-31)
  const int rbase = m0 + mi * 16 + ((lane < 16) ? 0 : 8);
  const int c0 = n0 + (ni2 * 2 + 0) * 16 + (lane & 15);
  const int c1 = c0 + 16;
#pragma unroll
  for (int r = 0; r < 8; ++r) {
    C[(size_t)(rbase + r) * ND3 + c0] = acc0[r];
    C[(size_t)(rbase + r) * ND3 + c1] = acc1[r];
  }
}

// ---------------------------------------------------------------------------
// Persistent sequential GRU scan. 64 blocks x 256 threads (8 wave32 waves).
// Block b owns hidden units d in [16*b, 16*b+16). Its 48x1024 fp32 slice of
// WhT (192 KB) is cached once in dynamic LDS (CDNA5: 320 KB/WGP), so every
// step's 48 length-1024 dot products run entirely out of LDS.
// One grid barrier per step (monotonic counter); h double-buffered.
// ---------------------------------------------------------------------------
static __device__ __forceinline__ float wave_reduce_sum(float s) {
#pragma unroll
  for (int o = 16; o >= 1; o >>= 1) s += __shfl_down(s, o, 32);
  return s;
}

__global__ __launch_bounds__(256)
void gru_scan_kernel(const float* __restrict__ xp,    // T x 3072 (x@Wi)
                     const int*   __restrict__ term,  // T
                     const float* __restrict__ WhT,   // 3072 x 1024
                     const float* __restrict__ bh,    // 3072
                     const float* __restrict__ h0,    // 1024 (initial state)
                     float* __restrict__ y,           // T x 1024 (layer outputs)
                     float* __restrict__ hfin,        // 1024 (final state out)
                     float* __restrict__ hbuf,        // 2 x 1024 (ping-pong)
                     unsigned* __restrict__ bar) {
  extern __shared__ float wslice[];                   // 48 * 1024 floats (192 KB)
  __shared__ float hs[DD];
  __shared__ float part[3][16];
  const int tid   = threadIdx.x;
  const int lane  = tid & 31;
  const int wave  = tid >> 5;
  const int nb    = gridDim.x;
  const int dbase = blockIdx.x * 16;

  // Cache this block's weight slice: row r = g*16+di  ->  WhT[g*1024 + dbase + di]
  for (int idx = tid; idx < 48 * DD; idx += 256) {
    const int r = idx >> 10;              // 0..47
    const int k = idx & (DD - 1);
    const int g = r >> 4, di = r & 15;
    wslice[idx] = WhT[(size_t)(g * DD + dbase + di) * DD + k];
  }
  __syncthreads();

  for (int t = 0; t < TT; ++t) {
    // Snapshot h (with reset-on-terminate) into LDS
    const float* hsrc = (t == 0) ? h0 : (hbuf + (size_t)(t & 1) * DD);
    const int tm = term[t];
    for (int k = tid; k < DD; k += 256) hs[k] = tm ? 0.0f : hsrc[k];
    if (tid == 0 && t + 1 < TT)
      __builtin_prefetch(xp + (size_t)(t + 1) * ND3 + dbase, 0, 1);
    __syncthreads();

    // 48 dots, one wave per dot (6 dots/wave), fully LDS-resident operands
#pragma unroll
    for (int i = 0; i < 6; ++i) {
      const int dotid = wave * 6 + i;     // 0..47 == g*16+di
      const float* w = wslice + (size_t)dotid * DD;
      float s = 0.0f;
      for (int k = lane; k < DD; k += 32) s += hs[k] * w[k];
      s = wave_reduce_sum(s);
      if (lane == 0) part[dotid >> 4][dotid & 15] = s;
    }
    __syncthreads();

    // Gate math + state update for this block's 16 units
    if (tid < 16) {
      const int d = dbase + tid;
      const float* xrow = xp + (size_t)t * ND3;
      const float hr = part[0][tid] + bh[d];
      const float hz = part[1][tid] + bh[d + DD];
      const float hn = part[2][tid] + bh[d + 2 * DD];
      const float r  = 1.0f / (1.0f + __expf(-(xrow[d] + hr)));
      const float z  = 1.0f / (1.0f + __expf(-(xrow[d + DD] + hz)));
      const float n  = tanhf(xrow[d + 2 * DD] + r * hn);
      const float hv = (1.0f - z) * n + z * hs[d];
      hbuf[(size_t)((t + 1) & 1) * DD + d] = hv;
      y[(size_t)t * DD + d] = hv;
      if (t == TT - 1) hfin[d] = hv;
    }

    // One grid barrier per step (monotonic counter => no reset race)
    __threadfence();
    __syncthreads();
    if (tid == 0) {
      atomicAdd(bar, 1u);
      const unsigned target = (unsigned)(t + 1) * (unsigned)nb;
      while (atomicAdd(bar, 0u) < target) __builtin_amdgcn_s_sleep(1);
    }
    __syncthreads();
  }
}

__global__ void init_bars_kernel(unsigned* bars, int n) {
  int i = threadIdx.x;
  if (i < n) bars[i] = 0u;
}

// ---------------------------------------------------------------------------
// Host launcher
// ---------------------------------------------------------------------------
extern "C" void kernel_launch(void* const* d_in, const int* in_sizes, int n_in,
                              void* d_out, int out_size, void* d_ws, size_t ws_size,
                              hipStream_t stream) {
  (void)in_sizes; (void)n_in; (void)out_size; (void)ws_size;

  const float* inputs      = (const float*)d_in[0];  // T x D
  const int*   term        = (const int*)  d_in[1];  // T
  const float* last_states = (const float*)d_in[2];  // L x D
  const float* Wi          = (const float*)d_in[3];  // L x D x 3D
  const float* Wh          = (const float*)d_in[4];  // L x D x 3D
  const float* bh          = (const float*)d_in[5];  // L x 3D
  float* out = (float*)d_out;                         // [T*D top-layer y | L*D finals]

  // Workspace carve-up
  char* ws = (char*)d_ws;
  size_t off = 0;
  unsigned short* WiT16 = (unsigned short*)(ws + off); off += (size_t)LL * DD * ND3 * sizeof(unsigned short);
  unsigned short* x16   = (unsigned short*)(ws + off); off += (size_t)TT * DD * sizeof(unsigned short);
  float* xp   = (float*)(ws + off); off += (size_t)TT * ND3 * sizeof(float);
  float* whT  = (float*)(ws + off); off += (size_t)ND3 * DD * sizeof(float);
  float* yA   = (float*)(ws + off); off += (size_t)TT * DD * sizeof(float);
  float* yB   = (float*)(ws + off); off += (size_t)TT * DD * sizeof(float);
  float* hbuf = (float*)(ws + off); off += (size_t)2 * DD * sizeof(float);
  unsigned* bars = (unsigned*)(ws + off); off += 4 * sizeof(unsigned);

  // Zero per-layer barrier counters (ws is poisoned; must re-init every call)
  init_bars_kernel<<<1, 32, 0, stream>>>(bars, LL);

  // Convert + transpose all input-projection weights once: WiT16[l][n][k]
  cvt_transpose_wi_kernel<<<dim3(ND3 / 32, DD / 32, LL), 256, 0, stream>>>(Wi, WiT16);

  const dim3 gemmGrid(ND3 / 64, TT / 64);   // 48 x 32
  const dim3 trGrid(ND3 / 32, DD / 32);     // 96 x 32
  const size_t scanLds = (size_t)48 * DD * sizeof(float);  // 192 KB / WGP

  for (int l = 0; l < LL; ++l) {
    const float* xin  = (l == 0) ? inputs : ((l & 1) ? yA : yB);
    float*       yout = (l == 3) ? out    : ((l & 1) ? yB : yA);

    // 1) layer input -> bf16
    {
      const int n = TT * DD;
      cvt_f32_bf16_kernel<<<(n + 255) / 256, 256, 0, stream>>>(xin, x16, n);
    }
    // 2) xp = x @ Wi[l]  (bf16 WMMA, fp32 accumulate)
    gemm_xp_kernel<<<gemmGrid, 256, 0, stream>>>(
        x16, WiT16 + (size_t)l * ND3 * DD, xp);
    // 3) WhT = Wh[l]^T for coalesced / conflict-free weight access
    transpose_wh_kernel<<<trGrid, 256, 0, stream>>>(
        Wh + (size_t)l * DD * ND3, whT);
    // 4) sequential scan (persistent, LDS-resident weights, 1 barrier/step)
    gru_scan_kernel<<<64, 256, scanLds, stream>>>(
        xp, term, whT, bh + (size_t)l * ND3, last_states + (size_t)l * DD,
        yout, out + (size_t)TT * DD + (size_t)l * DD, hbuf, bars + l);
  }
}
